// BGraphConvolution_28295244546730
// MI455X (gfx1250) — compile-verified, matrix operand order checked by reading, not run on hardware
//
#include <hip/hip_runtime.h>
#include <stdint.h>

#define IN_F 512
#define OUT_F 512
#define BJ_ITERS 10

typedef __attribute__((ext_vector_type(2)))  float   v2f;
typedef __attribute__((ext_vector_type(8)))  float   v8f;
typedef __attribute__((ext_vector_type(16))) __bf16  v16bf;

__device__ __forceinline__ unsigned short f32_to_bf16(float f) {
    union { float f; uint32_t u; } c; c.f = f;
    uint32_t u = c.u;
    // round-to-nearest-even
    return (unsigned short)((u + 0x7FFFu + ((u >> 16) & 1u)) >> 16);
}

// wt = weight^T / 512   (512x512)
__global__ void k_transpose_scale(const float* __restrict__ w, float* __restrict__ wt) {
    int idx = blockIdx.x * blockDim.x + threadIdx.x;
    int i = idx >> 9, j = idx & 511;
    wt[i * 512 + j] = w[j * 512 + i] * (1.0f / 512.0f);
}

// G = wt^T * wt  via V_WMMA_F32_16X16X4_F32. One 16x16 tile per wave.
__global__ __launch_bounds__(256) void k_gram_f32(const float* __restrict__ wt,
                                                  float* __restrict__ G) {
    int wave = blockIdx.x * 8 + (threadIdx.x >> 5);   // 1024 tiles total
    int lane = threadIdx.x & 31;
    int hi = lane >> 4, l = lane & 15;
    int m0 = (wave >> 5) * 16, n0 = (wave & 31) * 16;
    v8f c = {};
    for (int k0 = 0; k0 < 512; k0 += 4) {
        int ka = k0 + hi * 2;
        v2f a, b;
        a.x = wt[ka * 512 + m0 + l];        // A[m,k] = wt[k,m]
        a.y = wt[(ka + 1) * 512 + m0 + l];
        b.x = wt[ka * 512 + n0 + l];        // B[k,n] = wt[k,n]
        b.y = wt[(ka + 1) * 512 + n0 + l];
        c = __builtin_amdgcn_wmma_f32_16x16x4_f32(false, a, false, b,
                                                  (short)0, c, false, false);
    }
#pragma unroll
    for (int r = 0; r < 8; ++r)
        G[(m0 + r + hi * 8) * 512 + n0 + l] = c[r];
}

// out = 1.5*wt - 0.5*(wt*G)
__global__ __launch_bounds__(256) void k_update_f32(const float* __restrict__ wt,
                                                    const float* __restrict__ G,
                                                    float* __restrict__ out) {
    int wave = blockIdx.x * 8 + (threadIdx.x >> 5);
    int lane = threadIdx.x & 31;
    int hi = lane >> 4, l = lane & 15;
    int m0 = (wave >> 5) * 16, n0 = (wave & 31) * 16;
    v8f c = {};
    for (int k0 = 0; k0 < 512; k0 += 4) {
        int ka = k0 + hi * 2;
        v2f a, b;
        a.x = wt[(m0 + l) * 512 + ka];      // A[m,k] = wt[m,k]
        a.y = wt[(m0 + l) * 512 + ka + 1];
        b.x = G[ka * 512 + n0 + l];
        b.y = G[(ka + 1) * 512 + n0 + l];
        c = __builtin_amdgcn_wmma_f32_16x16x4_f32(false, a, false, b,
                                                  (short)0, c, false, false);
    }
#pragma unroll
    for (int r = 0; r < 8; ++r) {
        int m = m0 + r + hi * 8, n = n0 + l;
        out[m * 512 + n] = 1.5f * wt[m * 512 + n] - 0.5f * c[r];
    }
}

__global__ void k_to_bf16(const float* __restrict__ in, unsigned short* __restrict__ out,
                          int n) {
    int i = blockIdx.x * blockDim.x + threadIdx.x;
    if (i < n) out[i] = f32_to_bf16(in[i]);
}

// row_ptr[t] = lower_bound(rows, t); rows are sorted.
__global__ void k_rowptr(const int* __restrict__ rows, int* __restrict__ rp,
                         int nrows, int E) {
    int t = blockIdx.x * blockDim.x + threadIdx.x;
    if (t > nrows) return;
    int lo = 0, hi = E;
    while (lo < hi) { int mid = (lo + hi) >> 1; if (rows[mid] < t) lo = mid + 1; else hi = mid; }
    rp[t] = lo;
}

// support(bf16) = x(f32) @ ortho_w.  B buffer wB[n][k] bf16 (= wt_final rows).
// Block: 32 rows of x staged in LDS as bf16; 8 waves cover N=512; each wave
// computes a 32x64 tile as 2x4 WMMAs. All fragments load as 2x b128
// (fragment uints are two contiguous 16B runs per lane).
__global__ __launch_bounds__(256) void k_gemm_bf16(const float* __restrict__ x,
                                                   const unsigned short* __restrict__ wB,
                                                   unsigned short* __restrict__ support) {
    __shared__ __align__(16) unsigned short lds_A[32 * 512];   // 32 KB
    int m0 = blockIdx.x * 32;
    // cooperative stage: float4 read -> 4x bf16 -> uint2 (8B) LDS write
    for (int e4 = threadIdx.x; e4 < 32 * 128; e4 += 256) {
        int e = e4 * 4;
        int r = e >> 9, cidx = e & 511;
        float4 xf = *(const float4*)(x + (size_t)(m0 + r) * 512 + cidx);
        union { unsigned short us[4]; uint2 u2; } p;
        p.us[0] = f32_to_bf16(xf.x); p.us[1] = f32_to_bf16(xf.y);
        p.us[2] = f32_to_bf16(xf.z); p.us[3] = f32_to_bf16(xf.w);
        *(uint2*)(lds_A + r * 512 + cidx) = p.u2;
    }
    __syncthreads();

    const uint4* ldsq = (const uint4*)lds_A;
    const uint4* wq = (const uint4*)wB;
    int wave = threadIdx.x >> 5, lane = threadIdx.x & 31;
    int hi = lane >> 4, l = lane & 15;

    union frag { v16bf v; uint4 q[2]; };
    v8f acc[2][4] = {};
    for (int k0 = 0; k0 < 512; k0 += 32) {
        frag a[2];
#pragma unroll
        for (int mt = 0; mt < 2; ++mt) {
            int base = ((mt * 16 + l) * 256 + (k0 >> 1) + hi * 4) >> 2;
            a[mt].q[0] = ldsq[base];
            a[mt].q[1] = ldsq[base + 2];
        }
#pragma unroll
        for (int nt = 0; nt < 4; ++nt) {
            int n = wave * 64 + nt * 16 + l;
            int bbase = (n * 256 + (k0 >> 1) + hi * 4) >> 2;
            frag b;
            b.q[0] = wq[bbase];
            b.q[1] = wq[bbase + 2];
#pragma unroll
            for (int mt = 0; mt < 2; ++mt)
                acc[mt][nt] = __builtin_amdgcn_wmma_f32_16x16x32_bf16(
                    false, a[mt].v, false, b.v, (short)0, acc[mt][nt], false, false);
        }
    }
#pragma unroll
    for (int mt = 0; mt < 2; ++mt)
#pragma unroll
        for (int nt = 0; nt < 4; ++nt) {
            int n = wave * 64 + nt * 16 + l;
#pragma unroll
            for (int r = 0; r < 8; ++r) {
                int row = m0 + mt * 16 + r + hi * 8;
                support[(size_t)row * 512 + n] = f32_to_bf16(acc[mt][nt][r]);
            }
        }
}

// One block per destination row; edges staged through LDS; each thread owns
// one packed uint (2 bf16 features) -> 2 f32 accumulators. Unrolled x4 so four
// independent gathers are in flight; bf16 support (102MB) is L2-resident.
__global__ __launch_bounds__(256) void k_spmm(const uint32_t* __restrict__ support_u,
                                              const int* __restrict__ cols,
                                              const float* __restrict__ vals,
                                              const int* __restrict__ rp,
                                              float* __restrict__ out) {
    __shared__ int   s_col[256];
    __shared__ float s_val[256];
    int row = blockIdx.x;
    int lo = rp[row], hiE = rp[row + 1];
    int t = threadIdx.x;
    float a0 = 0.f, a1 = 0.f;
    for (int base = lo; base < hiE; base += 256) {
        int cnt = min(256, hiE - base);
        if (t < cnt) { s_col[t] = cols[base + t]; s_val[t] = vals[base + t]; }
        // prefetch next edge batch (global_prefetch_b8), 64B stride per lane
        if (base + 256 < hiE && t < 16) {
            __builtin_prefetch(&cols[base + 256 + t * 16], 0, 0);
            __builtin_prefetch(&vals[base + 256 + t * 16], 0, 0);
        }
        __syncthreads();
        int e = 0;
        for (; e + 4 <= cnt; e += 4) {
            int   c0 = s_col[e],     c1 = s_col[e + 1];
            int   c2 = s_col[e + 2], c3 = s_col[e + 3];
            float v0 = s_val[e],     v1 = s_val[e + 1];
            float v2 = s_val[e + 2], v3 = s_val[e + 3];
            uint32_t u0 = support_u[(size_t)c0 * 256 + t];
            uint32_t u1 = support_u[(size_t)c1 * 256 + t];
            uint32_t u2 = support_u[(size_t)c2 * 256 + t];
            uint32_t u3 = support_u[(size_t)c3 * 256 + t];
            a0 = fmaf(v0, __uint_as_float(u0 << 16), a0);
            a1 = fmaf(v0, __uint_as_float(u0 & 0xFFFF0000u), a1);
            a0 = fmaf(v1, __uint_as_float(u1 << 16), a0);
            a1 = fmaf(v1, __uint_as_float(u1 & 0xFFFF0000u), a1);
            a0 = fmaf(v2, __uint_as_float(u2 << 16), a0);
            a1 = fmaf(v2, __uint_as_float(u2 & 0xFFFF0000u), a1);
            a0 = fmaf(v3, __uint_as_float(u3 << 16), a0);
            a1 = fmaf(v3, __uint_as_float(u3 & 0xFFFF0000u), a1);
        }
        for (; e < cnt; ++e) {
            float v = s_val[e];
            uint32_t u = support_u[(size_t)s_col[e] * 256 + t];
            a0 = fmaf(v, __uint_as_float(u << 16), a0);
            a1 = fmaf(v, __uint_as_float(u & 0xFFFF0000u), a1);
        }
        __syncthreads();
    }
    out[(size_t)row * 512 + 2 * t]     = a0;
    out[(size_t)row * 512 + 2 * t + 1] = a1;
}

extern "C" void kernel_launch(void* const* d_in, const int* in_sizes, int n_in,
                              void* d_out, int out_size, void* d_ws, size_t ws_size,
                              hipStream_t stream) {
    const float* x        = (const float*)d_in[0];
    const float* weight   = (const float*)d_in[1];
    const int*   adj_rows = (const int*)d_in[2];
    const int*   adj_cols = (const int*)d_in[3];
    const float* adj_vals = (const float*)d_in[4];
    int N = in_sizes[0] / IN_F;     // 100000
    int E = in_sizes[2];            // 3200000
    float* out = (float*)d_out;

    char* ws = (char*)d_ws;
    size_t off = 0;
    unsigned short* support = (unsigned short*)(ws + off);
    off += (size_t)N * OUT_F * sizeof(unsigned short);
    off = (off + 255) & ~(size_t)255;
    float* wtA = (float*)(ws + off); off += 512 * 512 * sizeof(float);
    float* wtB = (float*)(ws + off); off += 512 * 512 * sizeof(float);
    float* G   = (float*)(ws + off); off += 512 * 512 * sizeof(float);
    unsigned short* wbf = (unsigned short*)(ws + off); off += 512 * 512 * sizeof(unsigned short);
    int* rp = (int*)(ws + off); off += (size_t)(N + 1) * sizeof(int);
    (void)ws_size; (void)n_in; (void)out_size;

    k_transpose_scale<<<1024, 256, 0, stream>>>(weight, wtA);
    float* cur = wtA; float* nxt = wtB;
    for (int it = 0; it < BJ_ITERS; ++it) {
        k_gram_f32<<<128, 256, 0, stream>>>(cur, G);
        k_update_f32<<<128, 256, 0, stream>>>(cur, G, nxt);
        float* tmp = cur; cur = nxt; nxt = tmp;
    }
    k_to_bf16<<<1024, 256, 0, stream>>>(cur, wbf, 512 * 512);
    k_rowptr<<<(N + 1 + 255) / 256, 256, 0, stream>>>(adj_rows, rp, N, E);
    k_gemm_bf16<<<N / 32, 256, 0, stream>>>(x, wbf, support);
    k_spmm<<<N, 256, 0, stream>>>((const uint32_t*)support, adj_cols, adj_vals, rp, out);
}
